// OTKernel_47253230190784
// MI455X (gfx1250) — compile-verified, matrix operand order checked by reading, not run on hardware
//
#include <hip/hip_runtime.h>

// ---------------------------------------------------------------- problem dims
#define BATCH    64
#define IN_SIZE  4096
#define IN_DIM   128
#define HEADS    2
#define OUT_SIZE 128
#define NM       (BATCH * HEADS)   // 128 independent sinkhorn problems
#define INV_EPS  10.0f             // 1/0.1
#define MAX_ITER 50

// ---------------------------------------------------------------- vector types
typedef __attribute__((ext_vector_type(16))) __bf16 v16bf;
typedef __attribute__((ext_vector_type(8)))  __bf16 v8bf;
typedef __attribute__((ext_vector_type(4)))  __bf16 v4bf;
typedef __attribute__((ext_vector_type(8)))  float  v8f;

// Load one 16-bit A/B fragment (16x32 A-layout, B mirrors it) for lane `hi` half.
// Per ISA: lanes<16 -> K = kbase+{0..7, 16..23}; lanes>=16 -> K = kbase+{8..15, 24..31}.
// Two contiguous 16-byte LDS reads per lane.
__device__ __forceinline__ v16bf frag_from_lds(const __bf16* rowp, int kbase, bool hi) {
    const int c0 = kbase + (hi ? 8 : 0);
    v8bf a = *(const v8bf*)(rowp + c0);
    v8bf b = *(const v8bf*)(rowp + c0 + 16);
    return __builtin_shufflevector(a, b, 0, 1, 2, 3, 4, 5, 6, 7,
                                         8, 9, 10, 11, 12, 13, 14, 15);
}

// ================================================================ phase 1
// K[nm][i][o] = bf16( exp( (x[n][i][:] . w[m][o][:]) / eps ) )
// Split-bf16 (hi/lo) 3-WMMA accumulation for ~fp32 dot precision.
// grid = (IN_SIZE/128, HEADS, BATCH), block = 256 (8 waves)
__global__ __launch_bounds__(256) void otk_gemm_exp(const float* __restrict__ x,
                                                    const float* __restrict__ w,
                                                    __bf16* __restrict__ Kmat) {
    const int itile = blockIdx.x, m = blockIdx.y, n = blockIdx.z;
    const int tid = threadIdx.x;
    const int lane = tid & 31, wave = tid >> 5;

    __shared__ __bf16 xh[128][128], xl[128][128];   // 32 KB + 32 KB
    __shared__ __bf16 wh[128][128], wl[128][128];   // 32 KB + 32 KB

    const float* xp = x + ((size_t)n * IN_SIZE + (size_t)itile * 128) * IN_DIM;
    const float* wp = w + (size_t)m * OUT_SIZE * IN_DIM;
    for (int t = tid; t < 128 * 128; t += 256) {
        const int r = t >> 7, c = t & 127;
        float xv = xp[t];
        __bf16 h = (__bf16)xv;
        xh[r][c] = h;
        xl[r][c] = (__bf16)(xv - (float)h);
        float wv = wp[t];
        __bf16 g = (__bf16)wv;
        wh[r][c] = g;
        wl[r][c] = (__bf16)(wv - (float)g);
    }
    __syncthreads();

    const bool hi = (lane >= 16);
    const int arow = wave * 16 + (lane & 15);

    v8f acc[8] = {};
    for (int kb = 0; kb < IN_DIM; kb += 32) {
        v16bf ah = frag_from_lds(&xh[arow][0], kb, hi);
        v16bf al = frag_from_lds(&xl[arow][0], kb, hi);
        for (int ot = 0; ot < 8; ++ot) {
            const int brow = ot * 16 + (lane & 15);
            v16bf bh = frag_from_lds(&wh[brow][0], kb, hi);
            v16bf bl = frag_from_lds(&wl[brow][0], kb, hi);
            acc[ot] = __builtin_amdgcn_wmma_f32_16x16x32_bf16(false, ah, false, bh,
                                                              (short)0, acc[ot], false, false);
            acc[ot] = __builtin_amdgcn_wmma_f32_16x16x32_bf16(false, ah, false, bl,
                                                              (short)0, acc[ot], false, false);
            acc[ot] = __builtin_amdgcn_wmma_f32_16x16x32_bf16(false, al, false, bh,
                                                              (short)0, acc[ot], false, false);
        }
    }

    // Epilogue: C vgpr r @ lane L -> i_loc = wave*16 + r + (L>=16 ? 8:0), o = ot*16 + L%16
    const size_t nm = (size_t)n * HEADS + m;
    __bf16* kout = Kmat + (nm * IN_SIZE + (size_t)itile * 128) * OUT_SIZE;
    for (int ot = 0; ot < 8; ++ot) {
        for (int r = 0; r < 8; ++r) {
            const int i_loc = wave * 16 + r + (hi ? 8 : 0);
            const int o     = ot * 16 + (lane & 15);
            kout[(size_t)i_loc * OUT_SIZE + o] = (__bf16)__expf(acc[ot][r] * INV_EPS);
        }
    }
}

// ================================================================ phase 2
// 50 Sinkhorn iterations per (n,m). Memory-bound GEMVs over L2-resident bf16 K;
// u (16 KB) and v live in LDS for the whole loop. grid = NM, block = 512 (16 waves).
__global__ __launch_bounds__(512) void otk_sinkhorn(const __bf16* __restrict__ Kmat,
                                                    float* __restrict__ uout,
                                                    float* __restrict__ vout) {
    const int nm = blockIdx.x;
    const int tid = threadIdx.x, lane = tid & 31, wave = tid >> 5;

    __shared__ float u[IN_SIZE];     // 16 KB
    __shared__ float v[OUT_SIZE];
    __shared__ float vsum[OUT_SIZE];

    const __bf16* Kb = Kmat + (size_t)nm * IN_SIZE * OUT_SIZE;
    for (int t = tid; t < OUT_SIZE; t += 512) v[t] = 1.0f;
    __syncthreads();

    const float a_marg = (float)OUT_SIZE / (float)IN_SIZE;   // 1/32

    for (int it = 0; it < MAX_ITER; ++it) {
        // ---- u_i = a / (K v)_i : one wave per row, lane covers 4 columns
        for (int i = wave; i < IN_SIZE; i += 16) {
            const __bf16* row = Kb + (size_t)i * OUT_SIZE + lane * 4;
            __builtin_prefetch(row + 16 * OUT_SIZE, 0, 1);   // global_prefetch_b8
            v4bf  kk = *(const v4bf*)row;
            float4 vv = *(const float4*)&v[lane * 4];
            float s = (float)kk[0] * vv.x + (float)kk[1] * vv.y +
                      (float)kk[2] * vv.z + (float)kk[3] * vv.w;
            #pragma unroll
            for (int off = 16; off >= 1; off >>= 1) s += __shfl_xor(s, off, 32);
            if (lane == 0) u[i] = a_marg / s;
        }
        __syncthreads();

        // ---- v_o = 1 / (u^T K)_o : lane owns 4 columns, waves split rows
        for (int t = tid; t < OUT_SIZE; t += 512) vsum[t] = 0.0f;
        __syncthreads();
        float a0 = 0.f, a1 = 0.f, a2 = 0.f, a3 = 0.f;
        for (int i = wave; i < IN_SIZE; i += 16) {
            v4bf kk = *(const v4bf*)(Kb + (size_t)i * OUT_SIZE + lane * 4);
            const float ui = u[i];
            a0 += ui * (float)kk[0];
            a1 += ui * (float)kk[1];
            a2 += ui * (float)kk[2];
            a3 += ui * (float)kk[3];
        }
        atomicAdd(&vsum[lane * 4 + 0], a0);   // ds_add_f32
        atomicAdd(&vsum[lane * 4 + 1], a1);
        atomicAdd(&vsum[lane * 4 + 2], a2);
        atomicAdd(&vsum[lane * 4 + 3], a3);
        __syncthreads();
        for (int t = tid; t < OUT_SIZE; t += 512) v[t] = 1.0f / vsum[t];
        __syncthreads();
    }

    for (int t = tid; t < IN_SIZE; t += 512)  uout[(size_t)nm * IN_SIZE + t]  = u[t];
    for (int t = tid; t < OUT_SIZE; t += 512) vout[(size_t)nm * OUT_SIZE + t] = v[t];
}

// ================================================================ phase 3
// out[n][o][m*128+d] = sum_i u_i*K[i][o]*v_o * x[n][i][d]   (TN GEMM, contract over i)
// Stage 32-i chunks transposed into LDS (fusing diag(u)*K*diag(v)), then WMMA.
// grid = NM, block = 256 (8 waves).
__global__ __launch_bounds__(256) void otk_output(const __bf16* __restrict__ Kmat,
                                                  const float* __restrict__ uu,
                                                  const float* __restrict__ vv,
                                                  const float* __restrict__ x,
                                                  float* __restrict__ out) {
    const int nm = blockIdx.x;
    const int n = nm / HEADS, m = nm % HEADS;
    const int tid = threadIdx.x, lane = tid & 31, wave = tid >> 5;

    // stride 40 halves = 80 B rows: keeps every 8-half chunk 16-B aligned, spreads banks
    __shared__ __bf16 Tt[OUT_SIZE][40];   // Tt[o][i_loc] = bf16(u_i * K[i][o] * v_o)
    __shared__ __bf16 xT[IN_DIM][40];     // xT[d][i_loc]
    __shared__ float  vsh[OUT_SIZE];

    const __bf16* Kb = Kmat + (size_t)nm * IN_SIZE * OUT_SIZE;
    const float*  ub = uu + (size_t)nm * IN_SIZE;
    const float*  xb = x + (size_t)n * IN_SIZE * IN_DIM;
    for (int t = tid; t < OUT_SIZE; t += 256) vsh[t] = vv[(size_t)nm * OUT_SIZE + t];

    const bool hi = (lane >= 16);
    const int arow = wave * 16 + (lane & 15);
    v8f acc[8] = {};

    for (int ic = 0; ic < IN_SIZE; ic += 32) {
        __syncthreads();   // previous chunk's fragment reads complete (and vsh ready)
        for (int t = tid; t < 32 * 128; t += 256) {
            const int r = t >> 7;          // i offset inside chunk
            const int c = t & 127;         // column (o for K, d for x) — coalesced reads
            const float tval = ub[ic + r] * (float)Kb[(size_t)(ic + r) * OUT_SIZE + c] * vsh[c];
            Tt[c][r] = (__bf16)tval;
            xT[c][r] = (__bf16)xb[(size_t)(ic + r) * IN_DIM + c];
        }
        __syncthreads();

        v16bf af = frag_from_lds(&Tt[arow][0], 0, hi);
        for (int dt = 0; dt < 8; ++dt) {
            v16bf bf_ = frag_from_lds(&xT[dt * 16 + (lane & 15)][0], 0, hi);
            acc[dt] = __builtin_amdgcn_wmma_f32_16x16x32_bf16(false, af, false, bf_,
                                                              (short)0, acc[dt], false, false);
        }
    }

    // C vgpr r @ lane L -> o = wave*16 + r + (L>=16?8:0), d = dt*16 + L%16
    float* ob = out + (size_t)n * OUT_SIZE * (HEADS * IN_DIM);
    for (int dt = 0; dt < 8; ++dt) {
        for (int r = 0; r < 8; ++r) {
            const int o = wave * 16 + r + (hi ? 8 : 0);
            const int d = dt * 16 + (lane & 15);
            ob[(size_t)o * (HEADS * IN_DIM) + m * IN_DIM + d] = acc[dt][r];
        }
    }
}

// ================================================================ launch
extern "C" void kernel_launch(void* const* d_in, const int* in_sizes, int n_in,
                              void* d_out, int out_size, void* d_ws, size_t ws_size,
                              hipStream_t stream) {
    (void)in_sizes; (void)n_in; (void)out_size; (void)ws_size;
    const float* x = (const float*)d_in[0];
    const float* w = (const float*)d_in[1];
    float* out = (float*)d_out;

    // workspace layout: K (bf16, 134.2 MB — L2-resident) | u (2 MB) | v (64 KB)
    char* ws = (char*)d_ws;
    __bf16* Kmat = (__bf16*)ws;
    const size_t k_bytes = (size_t)NM * IN_SIZE * OUT_SIZE * sizeof(__bf16);
    float* u = (float*)(ws + k_bytes);
    float* v = (float*)(ws + k_bytes + (size_t)NM * IN_SIZE * sizeof(float));

    dim3 g1(IN_SIZE / 128, HEADS, BATCH);
    otk_gemm_exp<<<g1, 256, 0, stream>>>(x, w, Kmat);
    otk_sinkhorn<<<NM, 512, 0, stream>>>(Kmat, u, v);
    otk_output<<<NM, 256, 0, stream>>>(Kmat, u, v, x, out);
}
